// UpwardCrossScaleFusion_5566277615870
// MI455X (gfx1250) — matrix operand order repaired and checked by reading, not compiled, not run on hardware
//
#include <hip/hip_runtime.h>
#include <hip/hip_bf16.h>

typedef __attribute__((ext_vector_type(16))) _Float16 v16h;
typedef __attribute__((ext_vector_type(8)))  _Float16 v8h;
typedef __attribute__((ext_vector_type(8)))  float    v8f;
typedef __attribute__((ext_vector_type(4)))  unsigned int u32x4;
typedef __attribute__((ext_vector_type(8)))  int i32x8;
typedef __attribute__((ext_vector_type(4)))  int i32x4;

// ---------------------------------------------------------------------------
// f32 -> f16 conversion (grid-stride)
// ---------------------------------------------------------------------------
__global__ __launch_bounds__(256) void cvt_f32_f16_kernel(
    const float* __restrict__ in, _Float16* __restrict__ out, size_t n) {
  size_t i = (size_t)blockIdx.x * blockDim.x + threadIdx.x;
  size_t stride = (size_t)gridDim.x * blockDim.x;
  for (; i < n; i += stride) out[i] = (_Float16)in[i];
}

// ---------------------------------------------------------------------------
// Generic WMMA GEMM:  C[M,N] = A[M,K] (f16) * W[N,K]^T (f16) + bias[N]
//   mode 0: C32[m,n] = acc + bias
//   mode 1: C16[m,n] = f16( gelu(acc + bias) )      (exact erf gelu)
// Tiling: block = 256 threads (8 waves), BM=64, BN=256, BK=32.
// Tile grid is 4x16 (16x16 tiles); each wave owns a 2(row) x 4(col) patch
// -> 8 WMMAs per K-step, 64 accum VGPRs/lane.
// A tile is staged global->LDS by the Tensor Data Mover (one tensor_load_to_lds
// per wave for its 8-row slice, TENSORcnt-tracked); B tile uses the vector path.
// A/B fragments are read from LDS in this code's contiguous-K order (same for A and B):
//   A 16x32 f16 : lane l -> m=l&15, kbase=(l>>4)*16 ; halves [kbase..+7],[kbase+8..+15]
//   B 32x16 f16 : lane l -> n=l&15, k=(l>>4)*16 + e, e=0..15 (contiguous in K)
// ---------------------------------------------------------------------------
__global__ __launch_bounds__(256) void gemm_f16_wmma_kernel(
    const _Float16* __restrict__ A, const _Float16* __restrict__ W,
    const float* __restrict__ bias, float* __restrict__ C32,
    _Float16* __restrict__ C16, int M, int N, int K, int mode) {
  constexpr int BM = 64, BN = 256, BK = 32;
  __shared__ __align__(64) _Float16 As[BM][BK];
  __shared__ __align__(64) _Float16 Bs[BN][BK];

  const int tid = threadIdx.x;
  const int lane = tid & 31;
  const int wv = tid >> 5;
  const int m0 = blockIdx.y * BM;
  const int n0 = blockIdx.x * BN;
  const int tr0 = (wv & 1) * 2;   // first of two 16-row stripes for this wave
  const int tc0 = (wv >> 1) * 4;  // first of four 16-col tiles for this wave

  // cooperative B load mapping: one 64B row (32 halves) per thread
  const int brow = tid;

  // fragment mappings (contiguous K, same order for A and B)
  const int mA = lane & 15;
  const int kA = (lane >> 4) * 16;
  const int nB = lane & 15;
  const int kB = (lane >> 4) * 16;

  // ---- TDM descriptor for this wave's 8-row A-tile slice (D# per ISA) ----
  // group1 (loop-invariant): data_size=1(2B); tensor_dim0=32, tensor_dim1=8;
  // tile_dim0=32, tile_dim1=8; tensor_dim0_stride=K elements.
  i32x8 g1;
  g1[0] = (1 << 16);                 // [17:16] data_size = 2 bytes
  g1[1] = (int)(32u << 16);          // [63:48] tensor_dim0 lo16 = 32
  g1[2] = (int)(8u << 16);           // [79:64] td0 hi16=0 | [95:80] td1 lo16=8
  g1[3] = (int)(32u << 16);          // [111:96] td1 hi16=0 | [127:112] tile_dim0=32
  g1[4] = 8;                         // [143:128] tile_dim1=8 | [159:144] tile_dim2=0
  g1[5] = K;                         // [207:160] tensor_dim0_stride lo32
  g1[6] = 0;                         // stride hi16 | dim1_stride lo16
  g1[7] = 0;
  const i32x4 gz4 = {0, 0, 0, 0};    // groups 2/3 unused (2-D tensor)
  const i32x8 gz8 = {0, 0, 0, 0, 0, 0, 0, 0};
  const unsigned lds_base = (unsigned)(size_t)(&As[wv * 8][0]);
  const unsigned long long grow =
      (unsigned long long)(size_t)(A + (size_t)(m0 + wv * 8) * K);

  v8f acc[2][4];
#pragma unroll
  for (int i = 0; i < 2; ++i)
#pragma unroll
    for (int j = 0; j < 4; ++j) acc[i][j] = (v8f){};

  for (int k0 = 0; k0 < K; k0 += BK) {
    if (k0 + BK < K)  // gfx1250 global_prefetch_b8 for next B tile
      __builtin_prefetch(&W[(size_t)(n0 + brow) * K + (k0 + BK)], 0, 1);

    // A tile: TDM async DMA (this wave's 8x32 slice), TENSORcnt-tracked
    {
      unsigned long long ga = grow + (unsigned long long)k0 * 2;
      u32x4 g0;
      g0[0] = 1u;                                   // count=1, user mode
      g0[1] = lds_base;                             // LDS byte address
      g0[2] = (unsigned)ga;                         // global addr lo
      g0[3] = (unsigned)(ga >> 32) | (2u << 30);    // global addr hi | type=2
      __builtin_amdgcn_tensor_load_to_lds(g0, g1, gz4, gz4, gz8, 0);
    }

    // B tile: vector path (256x32 halves, kept [n][k])
    const uint4* wp = (const uint4*)&W[(size_t)(n0 + brow) * K + k0];
#pragma unroll
    for (int q = 0; q < 4; ++q) *((uint4*)&Bs[brow][0] + q) = wp[q];

    __builtin_amdgcn_s_wait_tensorcnt(0);
    __syncthreads();

    v16h a[2];
#pragma unroll
    for (int i = 0; i < 2; ++i) {
      v8h alo = *(const v8h*)&As[(tr0 + i) * 16 + mA][kA];
      v8h ahi = *(const v8h*)&As[(tr0 + i) * 16 + mA][kA + 8];
      a[i] = __builtin_shufflevector(alo, ahi, 0, 1, 2, 3, 4, 5, 6, 7, 8, 9,
                                     10, 11, 12, 13, 14, 15);
    }
    v16h b[4];
#pragma unroll
    for (int j = 0; j < 4; ++j)
      b[j] = *(const v16h*)&Bs[(tc0 + j) * 16 + nB][kB];

#pragma unroll
    for (int i = 0; i < 2; ++i)
#pragma unroll
      for (int j = 0; j < 4; ++j)
        acc[i][j] = __builtin_amdgcn_wmma_f32_16x16x32_f16(
            false, a[i], false, b[j], (short)0, acc[i][j], false, false);
    __syncthreads();
  }

  // epilogue: D layout = lane l, reg r -> m = tile_m + (l>>4)*8 + r, n = l&15
#pragma unroll
  for (int i = 0; i < 2; ++i) {
    const int mbase = m0 + (tr0 + i) * 16 + (lane >> 4) * 8;
#pragma unroll
    for (int j = 0; j < 4; ++j) {
      const int n = n0 + (tc0 + j) * 16 + (lane & 15);
      const float bv = bias[n];
#pragma unroll
      for (int r = 0; r < 8; ++r) {
        float v = acc[i][j][r] + bv;
        size_t off = (size_t)(mbase + r) * N + n;
        if (mode == 0) {
          C32[off] = v;
        } else {
          float g = 0.5f * v * (1.0f + erff(v * 0.70710678118654752f));
          C16[off] = (_Float16)g;
        }
      }
    }
  }
}

// ---------------------------------------------------------------------------
// Banded cross-scale attention. One thread per (b, t, h). L = 2*band+1 (<=9).
// Band centers follow np.round(np.linspace(0, Tk-1, Tq)) (round-half-even).
// Writes msg directly as f16 (operand of the Wo GEMM).
// ---------------------------------------------------------------------------
__global__ __launch_bounds__(256) void attn_band_kernel(
    const float* __restrict__ Q, const float* __restrict__ Kb,
    const float* __restrict__ Vb, _Float16* __restrict__ msg16, int B, int Tq,
    int Tk, int H, int band) {
  const int D = 512;
  int g = blockIdx.x * blockDim.x + threadIdx.x;
  int total = B * Tq * H;
  if (g >= total) return;
  int h = g % H;
  int t = (g / H) % Tq;
  int b = g / (H * Tq);
  int hd = D / H;

  long center =
      (Tq > 1) ? llrint((double)t * (double)(Tk - 1) / (double)(Tq - 1)) : 0;
  const float scale = rsqrtf((float)hd);
  const float* qp = Q + ((size_t)(b * Tq + t) * D + h * hd);

  int L = 2 * band + 1;
  if (L > 9) L = 9;
  float sc[9];
  int kidx[9];
  for (int l = 0; l < L; ++l) {
    long kt = center - band + l;
    if (kt < 0) kt = 0;
    if (kt > Tk - 1) kt = Tk - 1;
    kidx[l] = (int)kt;
    const float* kp = Kb + ((size_t)(b * Tk + kt) * D + h * hd);
    float acc = 0.f;
    for (int e = 0; e < hd; ++e) acc = fmaf(qp[e], kp[e], acc);
    sc[l] = acc * scale;
  }
  float mx = sc[0];
  for (int l = 1; l < L; ++l) mx = fmaxf(mx, sc[l]);
  float ssum = 0.f;
  for (int l = 0; l < L; ++l) {
    sc[l] = expf(sc[l] - mx);
    ssum += sc[l];
  }
  float inv = 1.f / ssum;

  _Float16* mp = msg16 + ((size_t)(b * Tq + t) * D + h * hd);
  for (int e = 0; e < hd; ++e) {
    float acc = 0.f;
    for (int l = 0; l < L; ++l)
      acc = fmaf(sc[l], Vb[(size_t)(b * Tk + kidx[l]) * D + h * hd + e], acc);
    mp[e] = (_Float16)(acc * inv);
  }
}

// ---------------------------------------------------------------------------
// x = LayerNorm(res + y) * gamma + beta ; writes f32 (residual chain) and/or
// f16 (next GEMM operand). One 256-thread block per d=512 token row.
// ---------------------------------------------------------------------------
__global__ __launch_bounds__(256) void residual_ln_kernel(
    const float* __restrict__ res, const float* __restrict__ y,
    const float* __restrict__ gamma, const float* __restrict__ beta,
    float* __restrict__ out32, _Float16* __restrict__ out16) {
  const int D = 512;
  __shared__ float sred[8];
  const size_t base = (size_t)blockIdx.x * D;
  const int tid = threadIdx.x, lane = tid & 31, wv = tid >> 5;

  float v0 = res[base + tid] + y[base + tid];
  float v1 = res[base + tid + 256] + y[base + tid + 256];

  float s = v0 + v1;
  for (int o = 16; o > 0; o >>= 1) s += __shfl_down(s, o);
  if (lane == 0) sred[wv] = s;
  __syncthreads();
  if (tid == 0) {
    float a = 0.f;
    for (int i = 0; i < 8; ++i) a += sred[i];
    sred[0] = a;
  }
  __syncthreads();
  const float mean = sred[0] / (float)D;
  __syncthreads();

  float d0 = v0 - mean, d1 = v1 - mean;
  float q = d0 * d0 + d1 * d1;
  for (int o = 16; o > 0; o >>= 1) q += __shfl_down(q, o);
  if (lane == 0) sred[wv] = q;
  __syncthreads();
  if (tid == 0) {
    float a = 0.f;
    for (int i = 0; i < 8; ++i) a += sred[i];
    sred[0] = a;
  }
  __syncthreads();
  const float rstd = rsqrtf(sred[0] / (float)D + 1e-5f);

  float o0 = d0 * rstd * gamma[tid] + beta[tid];
  float o1 = d1 * rstd * gamma[tid + 256] + beta[tid + 256];
  if (out32) {
    out32[base + tid] = o0;
    out32[base + tid + 256] = o1;
  }
  if (out16) {
    out16[base + tid] = (_Float16)o0;
    out16[base + tid + 256] = (_Float16)o1;
  }
}

// ---------------------------------------------------------------------------
// Host orchestration
// ---------------------------------------------------------------------------
extern "C" void kernel_launch(void* const* d_in, const int* in_sizes, int n_in,
                              void* d_out, int out_size, void* d_ws,
                              size_t ws_size, hipStream_t stream) {
  (void)in_sizes; (void)n_in; (void)out_size; (void)ws_size;
  const float* seq0 = (const float*)d_in[0];
  const float* seq1 = (const float*)d_in[1];
  const float* seq2 = (const float*)d_in[2];
  const float* Wq = (const float*)d_in[3];
  const float* bq = (const float*)d_in[4];
  const float* Wk = (const float*)d_in[5];
  const float* bk = (const float*)d_in[6];
  const float* Wv = (const float*)d_in[7];
  const float* bv = (const float*)d_in[8];
  const float* Wo = (const float*)d_in[9];
  const float* bo = (const float*)d_in[10];
  const float* ln1g = (const float*)d_in[11];
  const float* ln1b = (const float*)d_in[12];
  const float* ln2g = (const float*)d_in[13];
  const float* ln2b = (const float*)d_in[14];
  const float* W1 = (const float*)d_in[15];
  const float* b1 = (const float*)d_in[16];
  const float* W2 = (const float*)d_in[17];
  const float* b2 = (const float*)d_in[18];
  // d_in[19] = nhead (==8), d_in[20] = band (==1): fixed instance, baked into
  // launch geometry (cannot be read host-side under graph capture).

  const int B = 8, T0 = 4096, T1 = 2048, T2 = 1024, D = 512, FF = 2048;
  const int H = 8, BAND = 1;
  const int M0 = B * T0, M1 = B * T1, M2 = B * T2;

  // ---- workspace layout (bump allocator, 256B aligned) ----
  char* p = (char*)d_ws;
  auto alloc = [&](size_t bytes) -> char* {
    char* r = p;
    p += (bytes + 255) & ~(size_t)255;
    return r;
  };
  _Float16* Wq16 = (_Float16*)alloc((size_t)D * D * 2);
  _Float16* Wk16 = (_Float16*)alloc((size_t)D * D * 2);
  _Float16* Wv16 = (_Float16*)alloc((size_t)D * D * 2);
  _Float16* Wo16 = (_Float16*)alloc((size_t)D * D * 2);
  _Float16* W116 = (_Float16*)alloc((size_t)FF * D * 2);
  _Float16* W216 = (_Float16*)alloc((size_t)D * FF * 2);
  _Float16* aq16 = (_Float16*)alloc((size_t)M0 * D * 2);   // q activations f16
  _Float16* akv16 = (_Float16*)alloc((size_t)M2 * D * 2);  // stage-1 kv f16
  float* RQ = (float*)alloc((size_t)M0 * D * 4);       // Q -> then O -> then h
  float* RKV = (float*)alloc((size_t)2 * M1 * D * 4);  // K|V -> then x32
  _Float16* m16 = (_Float16*)alloc((size_t)M0 * D * 2);   // msg16 -> then x16
  _Float16* g16 = (_Float16*)alloc((size_t)M0 * FF * 2);  // gelu(FFN1) f16
  _Float16* o16s1 = (_Float16*)alloc((size_t)M1 * D * 2);  // stage-1 out f16

  auto cvt = [&](const float* src, _Float16* dst, size_t n) {
    cvt_f32_f16_kernel<<<512, 256, 0, stream>>>(src, dst, n);
  };
  auto gemm = [&](const _Float16* Aa, const _Float16* Ww, const float* bb,
                  float* c32, _Float16* c16, int M, int N, int K, int mode) {
    dim3 grid(N / 256, M / 64);
    gemm_f16_wmma_kernel<<<grid, 256, 0, stream>>>(Aa, Ww, bb, c32, c16, M, N,
                                                   K, mode);
  };

  // ---- one-time weight conversion ----
  cvt(Wq, Wq16, (size_t)D * D);
  cvt(Wk, Wk16, (size_t)D * D);
  cvt(Wv, Wv16, (size_t)D * D);
  cvt(Wo, Wo16, (size_t)D * D);
  cvt(W1, W116, (size_t)FF * D);
  cvt(W2, W216, (size_t)D * FF);

  auto stage = [&](const float* qsrc, int Mq, int Tq, const _Float16* kv16,
                   int Mkv, int Tk, float* out32, _Float16* out16) {
    float* Qb = RQ;
    float* Kb = RKV;
    float* Vb = RKV + (size_t)Mkv * D;
    float* x32 = RKV;  // reuse after K/V consumed
    cvt(qsrc, aq16, (size_t)Mq * D);
    gemm(aq16, Wq16, bq, Qb, nullptr, Mq, D, D, 0);
    gemm(kv16, Wk16, bk, Kb, nullptr, Mkv, D, D, 0);
    gemm(kv16, Wv16, bv, Vb, nullptr, Mkv, D, D, 0);
    attn_band_kernel<<<(Mq * H + 255) / 256, 256, 0, stream>>>(
        Qb, Kb, Vb, m16, B, Tq, Tk, H, BAND);
    gemm(m16, Wo16, bo, RQ, nullptr, Mq, D, D, 0);  // O into RQ (Q dead)
    residual_ln_kernel<<<Mq, 256, 0, stream>>>(qsrc, RQ, ln1g, ln1b, x32,
                                               m16);  // x16 into m16 (dead)
    gemm(m16, W116, b1, nullptr, g16, Mq, FF, D, 1);  // fused gelu -> f16
    gemm(g16, W216, b2, RQ, nullptr, Mq, D, FF, 0);   // h into RQ (O dead)
    residual_ln_kernel<<<Mq, 256, 0, stream>>>(x32, RQ, ln2g, ln2b, out32,
                                               out16);
  };

  // stage 1: seq1 attends to seq2 -> updated seq1 (kept only as f16 operand)
  cvt(seq2, akv16, (size_t)M2 * D);
  stage(seq1, M1, T1, akv16, M2, T2, nullptr, o16s1);
  // stage 2: seq0 attends to updated seq1 -> final output (f32)
  stage(seq0, M0, T0, o16s1, M1, T1, (float*)d_out, nullptr);
}
